// _OneDimensionalSparseAttention_13915694039629
// MI455X (gfx1250) — matrix operand order repaired and checked by reading, not compiled, not run on hardware
//
#include <hip/hip_runtime.h>
#include <hip/hip_bf16.h>

// ---------------- problem constants ----------------
#define B_  2
#define H_  8
#define C_  2048
#define E_  1024
#define D_  64
#define KP_ 8
#define P_  16
#define HD_ 512          // H*D
#define M_  4096         // B*C

typedef __bf16 bf16;
typedef __attribute__((ext_vector_type(8)))  __bf16 v8bf;
typedef __attribute__((ext_vector_type(16))) __bf16 v16bf;
typedef __attribute__((ext_vector_type(8)))  float  v8f;
typedef __attribute__((ext_vector_type(4)))  float  v4f;

// ---------------- workspace layout (bytes) ----------------
#define OFF_XB   (size_t)0                 // x  bf16   4096x1024      8 MB
#define OFF_WKT  (size_t)8388608           // WkT bf16  512x1024       1 MB
#define OFF_WQT  (size_t)9437184
#define OFF_WVT  (size_t)10485760
#define OFF_WUT  (size_t)11534336          // WuT bf16 1024x512        1 MB
#define OFF_KT   (size_t)12582912          // Kt f32 (b,h,c,d)         8 MB
#define OFF_QT   (size_t)20971520
#define OFF_VT   (size_t)29360128
#define OFF_IDX  (size_t)37748736          // idx  i32 (b,h,c,P)       2 MB
#define OFF_NW   (size_t)39845888          // new_weights f32          2 MB
#define OFF_SUM  (size_t)41943040          // dens sums f32 (b,h,P,KP) 8 KB
#define OFF_UNI  (size_t)41951232          // united bf16 4096x512     4 MB

// ======================================================================
// 1) elementwise convert x -> bf16
// ======================================================================
__global__ __launch_bounds__(256) void k_cvt_x(const float* __restrict__ x,
                                               bf16* __restrict__ xb, int n) {
  int i = blockIdx.x * 256 + threadIdx.x;
  if (i < n) xb[i] = (bf16)x[i];
}

// transpose + convert: W (K x N, row major f32) -> Wt (N x K, row major bf16)
__global__ __launch_bounds__(256) void k_tw(const float* __restrict__ W,
                                            bf16* __restrict__ Wt, int K, int N) {
  int t = blockIdx.x * 256 + threadIdx.x;
  if (t < K * N) {
    int k = t % K, n = t / K;
    Wt[t] = (bf16)W[(size_t)k * N + n];
  }
}

__global__ __launch_bounds__(256) void k_zero_sums(float* __restrict__ sums) {
  int i = blockIdx.x * 256 + threadIdx.x;
  if (i < B_ * H_ * P_ * KP_) sums[i] = 0.f;
}

// ----------------------------------------------------------------------
// WMMA fragment helpers (per-lane layouts from CDNA5 ISA §7.12.2)
// A 16x32 bf16: lane<16 -> K {0..7,16..23}; lane>=16 -> K {8..15,24..31}
// B 32x16 bf16: lane holds column lane&15, K as for A (lane<16 {0..7,16..23}, else {8..15,24..31})
// ----------------------------------------------------------------------
__device__ __forceinline__ v16bf load_a_frag(const bf16* p) {
  v8bf a0 = *(const v8bf*)(p);
  v8bf a1 = *(const v8bf*)(p + 16);
  return __builtin_shufflevector(a0, a1, 0, 1, 2, 3, 4, 5, 6, 7,
                                 8, 9, 10, 11, 12, 13, 14, 15);
}

// epilogue store with (b,h,c,d) remap + scale
__device__ __forceinline__ void store_qkv_tile(float* __restrict__ dst,
                                               const v8f& acc, int mtile,
                                               int ntile, int lane, float scale) {
  const int n = ntile + (lane & 15);
  const int mb = mtile + ((lane >> 4) << 3);
  const int h = n >> 6, d = n & 63;
#pragma unroll
  for (int r = 0; r < 8; ++r) {
    int m = mb + r;
    int bb = m >> 11, c = m & (C_ - 1);
    dst[(((size_t)(bb * H_ + h) * C_ + c) * D_) + d] = acc[r] * scale;
  }
}

// ======================================================================
// 2) QKV projection GEMM: xb(4096x1024) @ WtT -> (b,h,c,d) f32
//    2x2 register-blocked WMMA: each wave owns a 32x32 tile (4 accums),
//    A/B fragments reused twice -> 16 FLOP per byte of L2 traffic.
//    Block = 8 waves (2x4) -> 64x128 tile.  blockIdx.z selects K/Q/V.
// ======================================================================
__global__ __launch_bounds__(256) void k_qkv_gemm(
    const bf16* __restrict__ xb,
    const bf16* __restrict__ WkT, const bf16* __restrict__ WqT,
    const bf16* __restrict__ WvT,
    float* __restrict__ Kt, float* __restrict__ Qt, float* __restrict__ Vt) {
  const int which = blockIdx.z;
  const bf16* Wt = (which == 0) ? WkT : (which == 1) ? WqT : WvT;
  float* dst = (which == 0) ? Kt : (which == 1) ? Qt : Vt;
  const float scale = (which == 2) ? 1.0f : 0.125f;   // 1/sqrt(64)

  const int lane = threadIdx.x & 31;
  const int wave = threadIdx.x >> 5;
  const int mtile = blockIdx.y * 64 + (wave >> 2) * 32;
  const int ntile = blockIdx.x * 128 + (wave & 3) * 32;

  const int ak0 = (lane >> 4) * 8;
  const bf16* ap0 = xb + (size_t)(mtile + (lane & 15)) * E_ + ak0;
  const bf16* ap1 = ap0 + (size_t)16 * E_;
  const int bk0 = (lane >> 4) * 8;
  const bf16* bp0 = Wt + (size_t)(ntile + (lane & 15)) * E_ + bk0;
  const bf16* bp1 = bp0 + (size_t)16 * E_;

  v8f acc00 = {}, acc01 = {}, acc10 = {}, acc11 = {};
  for (int kk = 0; kk < E_; kk += 32) {
    __builtin_prefetch(ap0 + kk + 256, 0, 1);
    __builtin_prefetch(ap1 + kk + 256, 0, 1);
    __builtin_prefetch(bp0 + kk + 256, 0, 1);
    __builtin_prefetch(bp1 + kk + 256, 0, 1);
    v16bf a0 = load_a_frag(ap0 + kk);
    v16bf a1 = load_a_frag(ap1 + kk);
    v16bf b0 = load_a_frag(bp0 + kk);
    v16bf b1 = load_a_frag(bp1 + kk);
    acc00 = __builtin_amdgcn_wmma_f32_16x16x32_bf16(false, a0, false, b0,
                                                    (short)0, acc00, false, false);
    acc01 = __builtin_amdgcn_wmma_f32_16x16x32_bf16(false, a0, false, b1,
                                                    (short)0, acc01, false, false);
    acc10 = __builtin_amdgcn_wmma_f32_16x16x32_bf16(false, a1, false, b0,
                                                    (short)0, acc10, false, false);
    acc11 = __builtin_amdgcn_wmma_f32_16x16x32_bf16(false, a1, false, b1,
                                                    (short)0, acc11, false, false);
  }

  store_qkv_tile(dst, acc00, mtile,      ntile,      lane, scale);
  store_qkv_tile(dst, acc01, mtile,      ntile + 16, lane, scale);
  store_qkv_tile(dst, acc10, mtile + 16, ntile,      lane, scale);
  store_qkv_tile(dst, acc11, mtile + 16, ntile + 16, lane, scale);
}

// ======================================================================
// 3) corner indices + density sums over context (reference axis=2 norm)
//    grid: (C/256, B*H); LDS reduction then global atomics
// ======================================================================
__global__ __launch_bounds__(256) void k_idx_dens(
    const float* __restrict__ means, const float* __restrict__ sigmas,
    int* __restrict__ idxb, float* __restrict__ sums) {
  __shared__ float s[P_ * KP_];
  const int bh = blockIdx.y;
  const int c = blockIdx.x * 256 + threadIdx.x;
  for (int i = threadIdx.x; i < P_ * KP_; i += 256) s[i] = 0.f;
  __syncthreads();

  const size_t base = ((size_t)bh * C_ + c) * KP_;
  float m[KP_], sg[KP_];
  int idx[P_];
#pragma unroll
  for (int k = 0; k < KP_; ++k) { m[k] = means[base + k]; sg[k] = sigmas[base + k]; }
#pragma unroll
  for (int k = 0; k < KP_; ++k) {
    int i0 = (int)floorf(m[k]);
    i0 = min(max(i0, 0), C_ - 1);
    idx[2 * k] = i0;
    idx[2 * k + 1] = min(i0 + 1, C_ - 1);
  }
  int* ip = idxb + ((size_t)bh * C_ + c) * P_;
#pragma unroll
  for (int p = 0; p < P_; ++p) ip[p] = idx[p];

#pragma unroll
  for (int p = 0; p < P_; ++p) {
    bool dup = false;
    for (int q = 0; q < p; ++q) dup |= (idx[q] == idx[p]);
    if (dup) continue;
    float pf = (float)idx[p];
#pragma unroll
    for (int k = 0; k < KP_; ++k) {
      float z = (pf - m[k]) / sg[k];
      atomicAdd(&s[p * KP_ + k], __expf(-0.5f * z * z));
    }
  }
  __syncthreads();
  if (threadIdx.x < P_ * KP_) atomicAdd(&sums[bh * P_ * KP_ + threadIdx.x], s[threadIdx.x]);
}

// ======================================================================
// 4) per-query core: weights, gather-dot, softmax, V-weighted head reps
//    one wave per (b,h,c); 8 waves per block
// ======================================================================
__global__ __launch_bounds__(256) void k_attn_core(
    const float* __restrict__ means, const float* __restrict__ sigmas,
    const float* __restrict__ values, const float* __restrict__ sums,
    const int* __restrict__ idxb,
    const float* __restrict__ Kt, const float* __restrict__ Qt,
    const float* __restrict__ Vt,
    float* __restrict__ nwb, bf16* __restrict__ united) {
  __shared__ float w_s[8][P_];
  __shared__ int i_s[8][P_];

  const int lane = threadIdx.x & 31;
  const int wave = threadIdx.x >> 5;
  const int q = blockIdx.x * 8 + wave;          // bh*C + c
  const int bh = q >> 11, c = q & (C_ - 1);
  const int bb = bh >> 3, h = bh & 7;

  const int pl = lane & 15;                      // lanes 16..31 mirror 0..15
  const int* ip = idxb + (size_t)q * P_;
  const int myidx = ip[pl];
  bool dup = false;
  for (int qq = 0; qq < P_; ++qq)
    if (qq < pl && ip[qq] == myidx) dup = true;

  // density-renormalized weight for point pl
  const size_t mbase = (size_t)q * KP_;
  float wsum = 0.f;
  if (!dup) {
    float pf = (float)myidx;
#pragma unroll
    for (int k = 0; k < KP_; ++k) {
      float mk = means[mbase + k];
      float z = (pf - mk) / sigmas[mbase + k];
      float dens = __expf(-0.5f * z * z);
      dens = dens / (sums[bh * P_ * KP_ + pl * KP_ + k] + 1e-8f);
      wsum += values[mbase + k] * dens;
    }
  }

  // dot(Q[c], K[idx])  (rows are 256B-aligned, v4f loads; hot in L2)
  const v4f* Qr = (const v4f*)(Qt + ((size_t)bh * C_ + c) * D_);
  const v4f* Kr = (const v4f*)(Kt + ((size_t)bh * C_ + myidx) * D_);
  float dot = 0.f;
#pragma unroll
  for (int i = 0; i < D_ / 4; ++i) {
    v4f a = Qr[i], b = Kr[i];
    dot += a[0] * b[0] + a[1] * b[1] + a[2] * b[2] + a[3] * b[3];
  }
  float t = wsum * dot;

  // softmax over the 16 points (xor reductions inside each half-wave)
  float mx = t;
#pragma unroll
  for (int m = 1; m < 16; m <<= 1) mx = fmaxf(mx, __shfl_xor(mx, m, 32));
  float e = __expf(t - mx);
  float es = e;
#pragma unroll
  for (int m = 1; m < 16; m <<= 1) es += __shfl_xor(es, m, 32);
  float w = e / es;

  if (lane < 16) {
    w_s[wave][pl] = w;
    i_s[wave][pl] = myidx;
    nwb[(size_t)q * P_ + pl] = w;
  }
  __syncthreads();

  // head_reps[d] = sum_p V[idx_p][d] * w_p ; each lane owns d = lane, lane+32
  float a0 = 0.f, a1 = 0.f;
#pragma unroll
  for (int p = 0; p < P_; ++p) {
    float vw = w_s[wave][p];
    const float* Vr = Vt + ((size_t)bh * C_ + i_s[wave][p]) * D_;
    a0 += Vr[lane] * vw;
    a1 += Vr[lane + 32] * vw;
  }
  bf16* u = united + ((size_t)(bb * C_ + c) * HD_) + h * D_;
  u[lane] = (bf16)a0;
  u[lane + 32] = (bf16)a1;
}

// ======================================================================
// 5) output projection: united(4096x512) @ Wu + bu -> new_out
//    same 2x2 register-blocked WMMA scheme; NT stores (write-once output)
// ======================================================================
__global__ __launch_bounds__(256) void k_out_gemm(
    const bf16* __restrict__ uni, const bf16* __restrict__ WuT,
    const float* __restrict__ bu, float* __restrict__ out) {
  const int lane = threadIdx.x & 31;
  const int wave = threadIdx.x >> 5;
  const int mtile = blockIdx.y * 64 + (wave >> 2) * 32;
  const int ntile = blockIdx.x * 128 + (wave & 3) * 32;

  const int ak0 = (lane >> 4) * 8;
  const bf16* ap0 = uni + (size_t)(mtile + (lane & 15)) * HD_ + ak0;
  const bf16* ap1 = ap0 + (size_t)16 * HD_;
  const int bk0 = (lane >> 4) * 8;
  const bf16* bp0 = WuT + (size_t)(ntile + (lane & 15)) * HD_ + bk0;
  const bf16* bp1 = bp0 + (size_t)16 * HD_;

  v8f acc00 = {}, acc01 = {}, acc10 = {}, acc11 = {};
  for (int kk = 0; kk < HD_; kk += 32) {
    __builtin_prefetch(ap0 + kk + 128, 0, 1);
    __builtin_prefetch(bp0 + kk + 128, 0, 1);
    v16bf a0 = load_a_frag(ap0 + kk);
    v16bf a1 = load_a_frag(ap1 + kk);
    v16bf b0 = load_a_frag(bp0 + kk);
    v16bf b1 = load_a_frag(bp1 + kk);
    acc00 = __builtin_amdgcn_wmma_f32_16x16x32_bf16(false, a0, false, b0,
                                                    (short)0, acc00, false, false);
    acc01 = __builtin_amdgcn_wmma_f32_16x16x32_bf16(false, a0, false, b1,
                                                    (short)0, acc01, false, false);
    acc10 = __builtin_amdgcn_wmma_f32_16x16x32_bf16(false, a1, false, b0,
                                                    (short)0, acc10, false, false);
    acc11 = __builtin_amdgcn_wmma_f32_16x16x32_bf16(false, a1, false, b1,
                                                    (short)0, acc11, false, false);
  }

  const int nlo = ntile + (lane & 15);
  const float bias0 = bu[nlo];
  const float bias1 = bu[nlo + 16];
  const v8f* accs[4] = {&acc00, &acc01, &acc10, &acc11};
#pragma unroll
  for (int t = 0; t < 4; ++t) {
    const int n = nlo + (t & 1) * 16;
    const int mb = mtile + (t >> 1) * 16 + ((lane >> 4) << 3);
    const float bias = (t & 1) ? bias1 : bias0;
#pragma unroll
    for (int r = 0; r < 8; ++r)
      __builtin_nontemporal_store((*accs[t])[r] + bias,
                                  out + (size_t)(mb + r) * E_ + n);
  }
}

// ======================================================================
// 6) dense attention matrix: one wave zero-fills its row (NT stores --
//    268MB write-once stream must not thrash L2), then scatters
//    dedup-merged softmax weights (in-order stores, no atomics needed)
// ======================================================================
__global__ __launch_bounds__(256) void k_attn_scatter(
    const int* __restrict__ idxb, const float* __restrict__ nwb,
    float* __restrict__ attn) {
  const int lane = threadIdx.x & 31;
  const int wave = threadIdx.x >> 5;
  const int q = blockIdx.x * 8 + wave;           // (b,h,c) flat
  float* row = attn + (size_t)q * C_;

  v4f z = {0.f, 0.f, 0.f, 0.f};
  v4f* row4 = (v4f*)row;
#pragma unroll
  for (int i = lane; i < C_ / 4; i += 32)
    __builtin_nontemporal_store(z, row4 + i);

  const int pl = lane & 15;
  const int* ip = idxb + (size_t)q * P_;
  const float* wp = nwb + (size_t)q * P_;
  const int myidx = ip[pl];
  float tot = 0.f;
  bool first = true;
#pragma unroll
  for (int qq = 0; qq < P_; ++qq) {
    int iq = ip[qq];
    if (iq == myidx) {
      if (qq < pl) first = false;
      tot += wp[qq];
    }
  }
  if (first && lane < 16) row[myidx] = tot;
}

// ======================================================================
extern "C" void kernel_launch(void* const* d_in, const int* in_sizes, int n_in,
                              void* d_out, int out_size, void* d_ws, size_t ws_size,
                              hipStream_t stream) {
  const float* x      = (const float*)d_in[0];
  // d_in[1] attention_mask: unused (matches reference)
  const float* means  = (const float*)d_in[2];
  const float* sigmas = (const float*)d_in[3];
  const float* values = (const float*)d_in[4];
  const float* Wk     = (const float*)d_in[5];
  const float* Wq     = (const float*)d_in[6];
  const float* Wv     = (const float*)d_in[7];
  const float* Wu     = (const float*)d_in[8];
  const float* bu     = (const float*)d_in[9];
  float* out = (float*)d_out;

  char* ws = (char*)d_ws;
  bf16*  xb   = (bf16*)(ws + OFF_XB);
  bf16*  WkT  = (bf16*)(ws + OFF_WKT);
  bf16*  WqT  = (bf16*)(ws + OFF_WQT);
  bf16*  WvT  = (bf16*)(ws + OFF_WVT);
  bf16*  WuT  = (bf16*)(ws + OFF_WUT);
  float* Kt   = (float*)(ws + OFF_KT);
  float* Qt   = (float*)(ws + OFF_QT);
  float* Vt   = (float*)(ws + OFF_VT);
  int*   idxb = (int*)(ws + OFF_IDX);
  float* nwb  = (float*)(ws + OFF_NW);
  float* sums = (float*)(ws + OFF_SUM);
  bf16*  uni  = (bf16*)(ws + OFF_UNI);

  float* new_out = out;                            // (B,C,E)
  float* attn    = out + (size_t)B_ * C_ * E_;     // (B,H,C,C)

  // ---- stage 1: conversions / transposes ----
  k_cvt_x<<<(M_ * E_) / 256, 256, 0, stream>>>(x, xb, M_ * E_);
  k_tw<<<(E_ * HD_) / 256, 256, 0, stream>>>(Wk, WkT, E_, HD_);
  k_tw<<<(E_ * HD_) / 256, 256, 0, stream>>>(Wq, WqT, E_, HD_);
  k_tw<<<(E_ * HD_) / 256, 256, 0, stream>>>(Wv, WvT, E_, HD_);
  k_tw<<<(HD_ * E_) / 256, 256, 0, stream>>>(Wu, WuT, HD_, E_);
  k_zero_sums<<<(B_ * H_ * P_ * KP_ + 255) / 256, 256, 0, stream>>>(sums);

  // ---- stage 2: QKV GEMMs on 2x2-blocked WMMA ----
  {
    dim3 grid(HD_ / 128, M_ / 64, 3);
    k_qkv_gemm<<<grid, 256, 0, stream>>>(xb, WkT, WqT, WvT, Kt, Qt, Vt);
  }

  // ---- stage 3: indices + context-normalization sums ----
  {
    dim3 grid(C_ / 256, B_ * H_);
    k_idx_dens<<<grid, 256, 0, stream>>>(means, sigmas, idxb, sums);
  }

  // ---- stage 4: per-query sparse attention core ----
  k_attn_core<<<(B_ * H_ * C_) / 8, 256, 0, stream>>>(
      means, sigmas, values, sums, idxb, Kt, Qt, Vt, nwb, uni);

  // ---- stage 5: output projection on 2x2-blocked WMMA ----
  {
    dim3 grid(E_ / 128, M_ / 64);
    k_out_gemm<<<grid, 256, 0, stream>>>(uni, WuT, bu, new_out);
  }

  // ---- stage 6: dense attention scatter (NT zero-fill) ----
  k_attn_scatter<<<(B_ * H_ * C_) / 8, 256, 0, stream>>>(idxb, nwb, attn);
}
